// TensorizedPC_65618510348797
// MI455X (gfx1250) — matrix operand order repaired and checked by reading, not compiled, not run on hardware
//
#include <hip/hip_runtime.h>

typedef __attribute__((ext_vector_type(16))) _Float16 v16h;
typedef __attribute__((ext_vector_type(8)))  _Float16 v8h;
typedef __attribute__((ext_vector_type(4)))  _Float16 v4h;
typedef __attribute__((ext_vector_type(8)))  float    v8f;

#define NVARS    2048
#define BATCH    256
#define KDIM     64
#define A_STRIDE 72   // f16 elems per LDS row (64 used, padded)
#define W_STRIDE 72
#define LOG2PI   1.8378770664093453f
#define LOG2E    1.44269504088896340736f   // 1/ln(2)
#define LN2      0.69314718055994530942f

// Raw hardware transcendentals: v_exp_f32 is 2^x, v_log_f32 is log2(x).
// Inputs here are guaranteed away from denormals (exp args <= 0, log args
// >= ~2^-6 since softmax weights >= ~1e-5 and the max exp term is 1.0),
// so we skip the libm denorm-guard fixups (v_cmp/v_cndmask/v_ldexp).
__device__ __forceinline__ float fast_exp(float x) {
    return __builtin_amdgcn_exp2f(x * LOG2E);
}
__device__ __forceinline__ float fast_log(float x) {
    return __builtin_amdgcn_logf(x) * LN2;
}

// ---------------------------------------------------------------------------
// Replicate np.random.RandomState(42) fold-index bookkeeping on device.
// MT19937 (rk_seed/rk_random/rk_interval) + Fisher-Yates shuffle, exactly as
// numpy's legacy RandomState.permutation(n). Single thread; runs once/launch.
// ---------------------------------------------------------------------------
__global__ void build_fold_indices_kernel(int* __restrict__ idx_out) {
    if (threadIdx.x != 0) return;
    __shared__ unsigned int mt[624];
    __shared__ int perm[NVARS];

    unsigned int s = 42u;
    for (int i = 0; i < 624; ++i) {
        mt[i] = s;
        s = 1812433253u * (s ^ (s >> 30)) + (unsigned)i + 1u;
    }
    int mti = 624;

    auto rk_random = [&]() -> unsigned int {
        if (mti >= 624) {
            for (int i = 0; i < 624; ++i) {
                unsigned int y = (mt[i] & 0x80000000u) | (mt[(i + 1) % 624] & 0x7fffffffu);
                mt[i] = mt[(i + 397) % 624] ^ (y >> 1) ^ ((y & 1u) ? 0x9908b0dfu : 0u);
            }
            mti = 0;
        }
        unsigned int y = mt[mti++];
        y ^= y >> 11;
        y ^= (y << 7)  & 0x9d2c5680u;
        y ^= (y << 15) & 0xefc60000u;
        y ^= y >> 18;
        return y;
    };
    auto rk_interval = [&](unsigned int mx) -> unsigned int {
        if (mx == 0u) return 0u;
        unsigned int mask = mx;
        mask |= mask >> 1; mask |= mask >> 2; mask |= mask >> 4;
        mask |= mask >> 8; mask |= mask >> 16;
        unsigned int v;
        while ((v = (rk_random() & mask)) > mx) { }
        return v;
    };

    int off = 0;
    for (int n = NVARS; n > 1; n >>= 1) {
        for (int i = 0; i < n; ++i) perm[i] = i;
        for (int i = n - 1; i > 0; --i) {
            unsigned int j = rk_interval((unsigned int)i);
            int tmp = perm[i]; perm[i] = perm[(int)j]; perm[(int)j] = tmp;
        }
        for (int i = 0; i < n; ++i) idx_out[off + i] = perm[i];
        off += n;
    }
}

// ---------------------------------------------------------------------------
// One block = one output fold. 256 threads = 8 waves (wave32).
// MODE 0: leaf layer (Gaussian log-density fused from x, mu).
// MODE 1: inner layer (reads fp32 log activations from prev buffer).
// Per arity: stage exp(logx - rowmax) f16 + weights f16 in LDS, do
// (256x64)@(64x64) with v_wmma_f32_16x16x32_f16, then log(y)+rowmax,
// accumulated over the two arities in registers.
// ---------------------------------------------------------------------------
template <int MODE>
__global__ __launch_bounds__(256, 1) void cp_layer_kernel(
    const float* __restrict__ src,   // MODE0: x (B,NVARS); MODE1: prev (Fp,B,64)
    const float* __restrict__ mu,    // MODE0 only: (NVARS,64)
    const float* __restrict__ wts,   // (F,2,64,64) softmax weights, fp32
    const int*   __restrict__ fold_idx, // (F,2)
    float*       __restrict__ out)   // (F,B,64) log activations
{
    __shared__ _Float16 sA[BATCH * A_STRIDE];
    __shared__ _Float16 sW[KDIM * W_STRIDE];
    __shared__ float    sMax[BATCH];

    const int f     = blockIdx.x;
    const int t     = threadIdx.x;
    const int lane  = t & 31;
    const int wv    = t >> 5;         // 8 waves
    const int lm    = lane & 15;
    const int lh    = lane >> 4;      // half-wave select per ISA layouts
    const int mwave = wv * 32;        // batch rows [mwave, mwave+32)

    v8f res[8];
    #pragma unroll
    for (int i = 0; i < 8; ++i)
        res[i] = (v8f){0.f,0.f,0.f,0.f,0.f,0.f,0.f,0.f};

    for (int a = 0; a < 2; ++a) {
        __syncthreads();   // previous iteration's LDS reads complete
        const int g = fold_idx[2 * f + a];

        // ---- stage A: one batch row per thread -------------------------
        float row[KDIM];
        if (MODE == 0) {
            const float xv = src[(size_t)t * NVARS + g];
            const float4* mp = (const float4*)(mu + (size_t)g * KDIM);
            #pragma unroll
            for (int i = 0; i < KDIM / 4; ++i) {
                float4 m4 = mp[i];
                float d0 = xv - m4.x, d1 = xv - m4.y, d2 = xv - m4.z, d3 = xv - m4.w;
                row[4*i+0] = -0.5f * d0 * d0 - 0.5f * LOG2PI;
                row[4*i+1] = -0.5f * d1 * d1 - 0.5f * LOG2PI;
                row[4*i+2] = -0.5f * d2 * d2 - 0.5f * LOG2PI;
                row[4*i+3] = -0.5f * d3 * d3 - 0.5f * LOG2PI;
            }
        } else {
            const float4* rp = (const float4*)(src + ((size_t)g * BATCH + t) * KDIM);
            #pragma unroll
            for (int i = 0; i < KDIM / 4; ++i) {
                float4 v4 = rp[i];
                row[4*i+0] = v4.x; row[4*i+1] = v4.y;
                row[4*i+2] = v4.z; row[4*i+3] = v4.w;
            }
        }
        float mx = row[0];
        #pragma unroll
        for (int k = 1; k < KDIM; ++k) mx = fmaxf(mx, row[k]);
        sMax[t] = mx;
        #pragma unroll
        for (int i = 0; i < KDIM / 4; ++i) {
            v4h h;
            h[0] = (_Float16)fast_exp(row[4*i+0] - mx);
            h[1] = (_Float16)fast_exp(row[4*i+1] - mx);
            h[2] = (_Float16)fast_exp(row[4*i+2] - mx);
            h[3] = (_Float16)fast_exp(row[4*i+3] - mx);
            *(v4h*)&sA[t * A_STRIDE + 4*i] = h;
        }

        // ---- stage W: (64,64) fp32 -> f16 in LDS -----------------------
        {
            const float4* wp = (const float4*)(wts + ((size_t)f * 2 + a) * (KDIM * KDIM));
            #pragma unroll
            for (int i = 0; i < 4; ++i) {
                int e4 = t + 256 * i;          // float4 index 0..1023
                float4 w4 = wp[e4];
                int lin = e4 * 4;
                int ki = lin >> 6, ko = lin & 63;
                v4h h;
                h[0] = (_Float16)w4.x; h[1] = (_Float16)w4.y;
                h[2] = (_Float16)w4.z; h[3] = (_Float16)w4.w;
                *(v4h*)&sW[ki * W_STRIDE + ko] = h;
            }
        }
        __syncthreads();

        // ---- WMMA: (256x64) @ (64x64), wave = 2 M-tiles x 4 N-tiles ----
        v8f ct[8];
        #pragma unroll
        for (int i = 0; i < 8; ++i)
            ct[i] = (v8f){0.f,0.f,0.f,0.f,0.f,0.f,0.f,0.f};

        #pragma unroll
        for (int kb = 0; kb < KDIM; kb += 32) {
            // B frags: lane n = lm, elems 0-7: K=kb+lh*8.., elems 8-15: K=kb+16+lh*8..
            v16h bf[4];
            #pragma unroll
            for (int nt = 0; nt < 4; ++nt) {
                const _Float16* p = &sW[(kb + lh * 8) * W_STRIDE + nt * 16 + lm];
                v16h b;
                #pragma unroll
                for (int s = 0; s < 16; ++s) b[s] = p[(s < 8 ? s : s + 8) * W_STRIDE];
                bf[nt] = b;
            }
            #pragma unroll
            for (int mt = 0; mt < 2; ++mt) {
                // A frag: lane m = lm, elems 0-7: K=kb+lh*8.., elems 8-15: K=kb+16+lh*8..
                const int m = mwave + mt * 16 + lm;
                const _Float16* ap = &sA[m * A_STRIDE + kb + lh * 8];
                v8h alo = *(const v8h*)ap;
                v8h ahi = *(const v8h*)(ap + 16);
                v16h af = __builtin_shufflevector(alo, ahi,
                    0,1,2,3,4,5,6,7,8,9,10,11,12,13,14,15);
                #pragma unroll
                for (int nt = 0; nt < 4; ++nt) {
                    ct[mt*4+nt] = __builtin_amdgcn_wmma_f32_16x16x32_f16(
                        false, af, false, bf[nt], (short)0, ct[mt*4+nt], false, false);
                }
            }
        }

        // ---- log(y) + rowmax, accumulate over arity --------------------
        #pragma unroll
        for (int mt = 0; mt < 2; ++mt) {
            #pragma unroll
            for (int r = 0; r < 8; ++r) {
                const float am = sMax[mwave + mt * 16 + lh * 8 + r];
                #pragma unroll
                for (int nt = 0; nt < 4; ++nt)
                    res[mt*4+nt][r] += fast_log(ct[mt*4+nt][r]) + am;
            }
        }
    }

    // ---- store (f, B, 64): half-wave writes 64B contiguous -------------
    #pragma unroll
    for (int mt = 0; mt < 2; ++mt) {
        #pragma unroll
        for (int nt = 0; nt < 4; ++nt) {
            #pragma unroll
            for (int r = 0; r < 8; ++r) {
                int brow = mwave + mt * 16 + lh * 8 + r;
                int n    = nt * 16 + lm;
                out[((size_t)f * BATCH + brow) * KDIM + n] = res[mt*4+nt][r];
            }
        }
    }
}

// ---------------------------------------------------------------------------
// Root layer: 2 folds -> 1 class. (256x64)@(64x1) per arity; scalar kernel.
// ---------------------------------------------------------------------------
__global__ __launch_bounds__(256) void root_kernel(
    const float* __restrict__ prev, const float* __restrict__ wts,
    const int* __restrict__ fold_idx, float* __restrict__ out)
{
    const int b = threadIdx.x;
    float res = 0.f;
    for (int a = 0; a < 2; ++a) {
        const int g = fold_idx[a];
        const float* row = prev + ((size_t)g * BATCH + b) * KDIM;
        float mx = row[0];
        #pragma unroll
        for (int k = 1; k < KDIM; ++k) mx = fmaxf(mx, row[k]);
        float s = 0.f;
        #pragma unroll
        for (int k = 0; k < KDIM; ++k)
            s += wts[a * KDIM + k] * fast_exp(row[k] - mx);
        res += fast_log(s) + mx;
    }
    out[b] = res;
}

// ---------------------------------------------------------------------------
extern "C" void kernel_launch(void* const* d_in, const int* in_sizes, int n_in,
                              void* d_out, int out_size, void* d_ws, size_t ws_size,
                              hipStream_t stream)
{
    (void)in_sizes; (void)out_size; (void)ws_size;
    const float* x  = (const float*)d_in[0];
    const float* mu = (const float*)d_in[1];

    // Weights: 11 layers, either as separate inputs d_in[2..12] or concatenated.
    const float* wl[11];
    {
        const float* wcat = (const float*)d_in[2];
        size_t woff = 0;
        for (int li = 0; li < 11; ++li) {
            wl[li] = (n_in >= 13) ? (const float*)d_in[2 + li] : (wcat + woff);
            woff += (li < 10) ? ((size_t)(1024 >> li) * 2 * 64 * 64) : (size_t)128;
        }
    }

    // Workspace carve-up: indices (16KB) + fp32 activation ping-pong.
    int*   idx  = (int*)d_ws;
    float* bufA = (float*)((char*)d_ws + 16384);                 // up to 1024 folds
    float* bufB = bufA + (size_t)1024 * BATCH * KDIM;            // up to 512 folds

    build_fold_indices_kernel<<<1, 32, 0, stream>>>(idx);

    // Cumulative index offsets per layer (layer li consumes 2048>>li ints).
    static const int idx_off[11] =
        {0, 2048, 3072, 3584, 3840, 3968, 4032, 4064, 4080, 4088, 4092};

    // Layer 0: fused Gaussian leaves, 2048 vars -> 1024 folds.
    cp_layer_kernel<0><<<1024, 256, 0, stream>>>(x, mu, wl[0], idx, bufA);

    // Layers 1..9: 512 -> 2 folds.
    float* cur = bufA; float* nxt = bufB;
    for (int li = 1; li <= 9; ++li) {
        int F = 1024 >> li;
        cp_layer_kernel<1><<<F, 256, 0, stream>>>(
            cur, nullptr, wl[li], idx + idx_off[li], nxt);
        float* tmp = cur; cur = nxt; nxt = tmp;
    }

    // Layer 10 (root): 2 folds -> (B, 1).
    root_kernel<<<1, 256, 0, stream>>>(cur, wl[10], idx + idx_off[10], (float*)d_out);
}
